// HKLinear_29128468201622
// MI455X (gfx1250) — matrix operand order repaired and checked by reading, not compiled, not run on hardware
//
#include <hip/hip_runtime.h>

// ---- problem constants (match reference) ----
#define IN_F   4096
#define OUT_F  4096
#define NCLUST 64
#define NQ     8192           // BATCH*SEQ = 4*2048
#define TEMP_INV 10.0f        // 1/TEMPERATURE
#define THRESH   0.01f

// ---- vector types ----
typedef __attribute__((ext_vector_type(16))) __bf16         bf16x16;
typedef __attribute__((ext_vector_type(8)))  float          f32x8;
typedef __attribute__((ext_vector_type(16))) unsigned short u16x16;
typedef __attribute__((ext_vector_type(8)))  unsigned short u16x8;
typedef __attribute__((ext_vector_type(4)))  unsigned short u16x4;

// =====================================================================
// Kernel 0: zero the global cluster mask
// =====================================================================
__global__ __launch_bounds__(64) void init_csel(int* csel) {
    csel[threadIdx.x] = 0;
}

// =====================================================================
// Kernel 1: routing. 1 block per query, 64 threads = 64 clusters.
//   qsel[q]  = any(softmax(x@C.T/T)[q,:] > thr)
//   csel[c] |= any over queries
// =====================================================================
__global__ __launch_bounds__(64) void route_kernel(
    const float* __restrict__ X, const float* __restrict__ C,
    int* __restrict__ qsel, int* __restrict__ csel)
{
    const int q = blockIdx.x;
    const int t = threadIdx.x;

    __shared__ float xs[IN_F];
    __shared__ float red[NCLUST];
    __shared__ int   ired[NCLUST];

    // stage x row into LDS (float4)
    const float4* xv = (const float4*)(X + (size_t)q * IN_F);
    for (int i = t; i < IN_F / 4; i += 64) ((float4*)xs)[i] = xv[i];
    __syncthreads();

    // dot(x, centroid[t])
    const float* c = C + (size_t)t * IN_F;
    float acc = 0.0f;
    for (int k = 0; k < IN_F; k += 4) {
        float4 cv = *(const float4*)(c + k);
        acc += xs[k] * cv.x + xs[k + 1] * cv.y + xs[k + 2] * cv.z + xs[k + 3] * cv.w;
    }
    float logit = acc * TEMP_INV;

    // softmax over the 64 threads
    red[t] = logit; __syncthreads();
    for (int s = 32; s > 0; s >>= 1) { if (t < s) red[t] = fmaxf(red[t], red[t + s]); __syncthreads(); }
    float m = red[0]; __syncthreads();
    float e = __expf(logit - m);
    red[t] = e; __syncthreads();
    for (int s = 32; s > 0; s >>= 1) { if (t < s) red[t] += red[t + s]; __syncthreads(); }
    float sum = red[0];

    int hit = ((e / sum) > THRESH) ? 1 : 0;
    ired[t] = hit; __syncthreads();
    for (int s = 32; s > 0; s >>= 1) { if (t < s) ired[t] |= ired[t + s]; __syncthreads(); }
    if (t == 0) qsel[q] = ired[0];
    if (hit) atomicOr(&csel[t], 1);
}

// =====================================================================
// Kernel 2: row_sel[r] = cluster_sel[assignments[r]]
// =====================================================================
__global__ __launch_bounds__(256) void rowsel_kernel(
    const int* __restrict__ assign, const int* __restrict__ csel,
    int* __restrict__ rsel)
{
    int r = blockIdx.x * 256 + threadIdx.x;
    rsel[r] = csel[assign[r]] ? 1 : 0;
}

// =====================================================================
// Kernel 3: masked GEMM  out = mask ? (X @ W.T + b) : 0
// bf16x3 split (hi*hi + hi*lo + lo*hi) with f32 WMMA accumulation.
// Block tile 256x128x32; 8 waves (4x2), each wave a 64x64 macro-tile
// (4x4 WMMA tiles) -> 48 WMMA per 32 ds_load_b128 per K-step.
// =====================================================================
#define BM 256
#define BN 128
#define BK 32
#define LDP 40   // padded LDS row stride in halfwords (80B, 16B-aligned)

__device__ __forceinline__ void split_bf16(float x, unsigned short& h, unsigned short& l) {
    unsigned int b = __float_as_uint(x);
    h = (unsigned short)(b >> 16);                       // truncate hi
    float hf = __uint_as_float((unsigned int)h << 16);
    float r  = x - hf;                                   // exact residual
    l = (unsigned short)(__float_as_uint(r) >> 16);      // truncate lo
}

// A fragment 16x32 bf16 per ISA layout: lane = (half<<4)|m,
// V0..3 = K{half*8 .. half*8+7}, V4..7 = K{16+half*8 ..}
__device__ __forceinline__ bf16x16 load_fragA(const unsigned short* s, int mBase, int lane) {
    int ml = lane & 15, h = lane >> 4;
    const unsigned short* p = s + (mBase + ml) * LDP;
    u16x8 x0 = *(const u16x8*)(p + h * 8);
    u16x8 x1 = *(const u16x8*)(p + 16 + h * 8);
    u16x16 r;
#pragma unroll
    for (int i = 0; i < 8; ++i) { r[i] = x0[i]; r[i + 8] = x1[i]; }
    return __builtin_bit_cast(bf16x16, r);
}

// B fragment 32x16 bf16: lane = (kh<<4)|n, V0..3 = K{kh*8 .. kh*8+7}, V4..7 = K{16+kh*8 ..}
__device__ __forceinline__ bf16x16 load_fragB(const unsigned short* s, int nBase, int lane) {
    int nl = lane & 15, kh = lane >> 4;
    const unsigned short* p = s + (nBase + nl) * LDP;
    u16x8 x0 = *(const u16x8*)(p + kh * 8);
    u16x8 x1 = *(const u16x8*)(p + 16 + kh * 8);
    u16x16 r;
#pragma unroll
    for (int i = 0; i < 8; ++i) { r[i] = x0[i]; r[i + 8] = x1[i]; }
    return __builtin_bit_cast(bf16x16, r);
}

__global__ __launch_bounds__(256) void gemm_kernel(
    const float* __restrict__ X, const float* __restrict__ W,
    const float* __restrict__ bias,
    const int* __restrict__ qsel, const int* __restrict__ rsel,
    float* __restrict__ out)
{
    __shared__ unsigned short sA[2][BM * LDP];   // [hi/lo][m][k]  2x20480B
    __shared__ unsigned short sB[2][BN * LDP];   // [hi/lo][n][k]  2x10240B

    const int t     = threadIdx.x;
    const int lane  = t & 31;
    const int wave  = t >> 5;
    const int waveM = wave >> 1;        // 0..3  (64 rows each)
    const int waveN = wave & 1;         // 0..1  (64 cols each)
    const int qBase = blockIdx.y * BM;
    const int rBase = blockIdx.x * BN;

    f32x8 acc[4][4];
#pragma unroll
    for (int mi = 0; mi < 4; ++mi)
#pragma unroll
        for (int ni = 0; ni < 4; ++ni)
#pragma unroll
            for (int v = 0; v < 8; ++v) acc[mi][ni][v] = 0.0f;

    float4 aR[8], bR[4];

    // prologue: global load of K-step 0
#pragma unroll
    for (int i = 0; i < 8; ++i) {
        int lin = t + i * 256, row = lin >> 3, cv = lin & 7;
        aR[i] = *(const float4*)(X + (size_t)(qBase + row) * IN_F + cv * 4);
    }
#pragma unroll
    for (int i = 0; i < 4; ++i) {
        int lin = t + i * 256, row = lin >> 3, cv = lin & 7;
        bR[i] = *(const float4*)(W + (size_t)(rBase + row) * IN_F + cv * 4);
    }

    const int KSTEPS = IN_F / BK;  // 128
    for (int kt = 0; kt < KSTEPS; ++kt) {
        // regs -> LDS (split into hi/lo bf16)
#pragma unroll
        for (int i = 0; i < 8; ++i) {
            int lin = t + i * 256, row = lin >> 3, cv = lin & 7;
            int base = row * LDP + cv * 4;
            u16x4 ah, al;
            const float* af = &aR[i].x;
#pragma unroll
            for (int j = 0; j < 4; ++j) { unsigned short h, l; split_bf16(af[j], h, l); ah[j] = h; al[j] = l; }
            *(u16x4*)&sA[0][base] = ah; *(u16x4*)&sA[1][base] = al;
        }
#pragma unroll
        for (int i = 0; i < 4; ++i) {
            int lin = t + i * 256, row = lin >> 3, cv = lin & 7;
            int base = row * LDP + cv * 4;
            u16x4 bh, bl;
            const float* bf = &bR[i].x;
#pragma unroll
            for (int j = 0; j < 4; ++j) { unsigned short h, l; split_bf16(bf[j], h, l); bh[j] = h; bl[j] = l; }
            *(u16x4*)&sB[0][base] = bh; *(u16x4*)&sB[1][base] = bl;
        }
        __syncthreads();

        // issue next K-step's global loads early to overlap with WMMA
        if (kt + 1 < KSTEPS) {
            int k0 = (kt + 1) * BK;
#pragma unroll
            for (int i = 0; i < 8; ++i) {
                int lin = t + i * 256, row = lin >> 3, cv = lin & 7;
                aR[i] = *(const float4*)(X + (size_t)(qBase + row) * IN_F + k0 + cv * 4);
            }
#pragma unroll
            for (int i = 0; i < 4; ++i) {
                int lin = t + i * 256, row = lin >> 3, cv = lin & 7;
                bR[i] = *(const float4*)(W + (size_t)(rBase + row) * IN_F + k0 + cv * 4);
            }
        }

        // B fragments resident across mi-loop; A fragments short-lived
        bf16x16 bF[4][2];
#pragma unroll
        for (int ni = 0; ni < 4; ++ni) {
            bF[ni][0] = load_fragB(&sB[0][0], waveN * 64 + ni * 16, lane);
            bF[ni][1] = load_fragB(&sB[1][0], waveN * 64 + ni * 16, lane);
        }
#pragma unroll
        for (int mi = 0; mi < 4; ++mi) {
            bf16x16 aH = load_fragA(&sA[0][0], waveM * 64 + mi * 16, lane);
            bf16x16 aL = load_fragA(&sA[1][0], waveM * 64 + mi * 16, lane);
#pragma unroll
            for (int ni = 0; ni < 4; ++ni) {
                acc[mi][ni] = __builtin_amdgcn_wmma_f32_16x16x32_bf16(
                    false, aH, false, bF[ni][0], (short)0, acc[mi][ni], false, false);
                acc[mi][ni] = __builtin_amdgcn_wmma_f32_16x16x32_bf16(
                    false, aH, false, bF[ni][1], (short)0, acc[mi][ni], false, false);
                acc[mi][ni] = __builtin_amdgcn_wmma_f32_16x16x32_bf16(
                    false, aL, false, bF[ni][0], (short)0, acc[mi][ni], false, false);
            }
        }
        __syncthreads();   // before next iteration overwrites LDS
    }

    // epilogue: bias + routing mask, write all elements (f32 C layout:
    // lane 0-15 -> N=lane, M=v; lane 16-31 -> N=lane-16, M=v+8)
    const int half = lane >> 4, nl = lane & 15;
#pragma unroll
    for (int mi = 0; mi < 4; ++mi) {
#pragma unroll
        for (int ni = 0; ni < 4; ++ni) {
            int rr = rBase + waveN * 64 + ni * 16 + nl;
            int rs = rsel[rr];
            float bv = bias[rr];
#pragma unroll
            for (int v = 0; v < 8; ++v) {
                int qq = qBase + waveM * 64 + mi * 16 + v + half * 8;
                bool on = (qsel[qq] != 0) && (rs != 0);
                out[(size_t)qq * OUT_F + rr] = on ? (acc[mi][ni][v] + bv) : 0.0f;
            }
        }
    }
}

// =====================================================================
// launch
// =====================================================================
extern "C" void kernel_launch(void* const* d_in, const int* in_sizes, int n_in,
                              void* d_out, int out_size, void* d_ws, size_t ws_size,
                              hipStream_t stream) {
    const float* X      = (const float*)d_in[0];
    const float* W      = (const float*)d_in[1];
    const float* bias   = (const float*)d_in[2];
    const float* cent   = (const float*)d_in[3];
    const int*   assign = (const int*)d_in[4];
    float* out = (float*)d_out;

    int* qsel = (int*)d_ws;            // NQ ints
    int* csel = qsel + NQ;             // NCLUST ints
    int* rsel = csel + NCLUST;         // OUT_F ints

    init_csel<<<1, 64, 0, stream>>>(csel);
    route_kernel<<<NQ, 64, 0, stream>>>(X, cent, qsel, csel);
    rowsel_kernel<<<OUT_F / 256, 256, 0, stream>>>(assign, csel, rsel);

    dim3 grid(OUT_F / BN, NQ / BM);    // (32, 32)
    gemm_kernel<<<grid, 256, 0, stream>>>(X, W, bias, qsel, rsel, out);
}
